// AscendDeepseekV2Attention_3745211482367
// MI455X (gfx1250) — matrix operand order repaired and checked by reading, not compiled, not run on hardware
//
#include <hip/hip_runtime.h>

// ---------------------------------------------------------------------------
// Types for CDNA5 WMMA (wave32, v_wmma_f32_16x16x32_bf16)
// ---------------------------------------------------------------------------
typedef __attribute__((ext_vector_type(16))) __bf16 bf16x16;
typedef __attribute__((ext_vector_type(8)))  float  f32x8;
typedef __attribute__((ext_vector_type(4)))  unsigned int u32x4;
typedef __attribute__((ext_vector_type(8)))  int i32x8;
typedef __attribute__((ext_vector_type(4)))  int i32x4;

union Frag {
    bf16x16        v;
    unsigned int   u[8];
    unsigned short h[16];
};

#if defined(__has_builtin)
#if __has_builtin(__builtin_amdgcn_tensor_load_to_lds) && \
    __has_builtin(__builtin_amdgcn_s_wait_tensorcnt)
#define HAVE_TDM 1
#endif
#endif
#ifndef HAVE_TDM
#define HAVE_TDM 0
#endif

__device__ __forceinline__ unsigned short f32_to_bf16(float f) {
    unsigned int u = __float_as_uint(f);
    unsigned int rnd = 0x7FFFu + ((u >> 16) & 1u);
    return (unsigned short)((u + rnd) >> 16);
}

#if HAVE_TDM
// ---------------------------------------------------------------------------
// Tensor Data Mover: 2D bf16 tile load global -> LDS.
// D# per CDNA5 ISA 8.3/8.4: group0 = {count|flags, lds_addr, gaddr_lo,
// gaddr_hi|type=2}; group1 = {data_size=2B, tensor_dim0/1, tile_dim0/1,
// tensor_dim0_stride}; groups 2/3 + extra group zero (<=2D tensor).
// clang-23 form: 6 args (g0 u32x4, g1 i32x8, g2 i32x4, g3 i32x4, i32x8, cpol)
// ---------------------------------------------------------------------------
__device__ __forceinline__ void tdm_load_2d(unsigned lds_off,
                                            const unsigned short* gptr,
                                            unsigned tile_w, unsigned tile_h,
                                            unsigned tensor_w, unsigned tensor_h,
                                            unsigned long long stride0_elems) {
    unsigned long long ga = (unsigned long long)gptr;
    u32x4 g0;
    g0.x = 1u;                                               // count = 1
    g0.y = lds_off;                                          // lds_addr (bytes)
    g0.z = (unsigned)ga;                                     // global_addr[31:0]
    g0.w = (unsigned)((ga >> 32) & 0x1FFFFFFu) | 0x80000000u; // addr[56:32] | type=2
    i32x8 g1;
    g1[0] = 0x10000;                                          // data_size = 1 (2 bytes)
    g1[1] = (int)((tensor_w & 0xFFFFu) << 16);                // tensor_dim0[15:0]
    g1[2] = (int)((tensor_w >> 16) | ((tensor_h & 0xFFFFu) << 16));
    g1[3] = (int)((tensor_h >> 16) | (tile_w << 16));         // tile_dim0
    g1[4] = (int)tile_h;                                      // tile_dim1 (tile_dim2=0)
    g1[5] = (int)(unsigned)stride0_elems;                     // dim0 stride lo
    g1[6] = (int)(unsigned)(stride0_elems >> 32);             // dim0 stride hi
    g1[7] = 0;
    i32x4 z4 = {0, 0, 0, 0};
    i32x8 z8 = {0, 0, 0, 0, 0, 0, 0, 0};
    __builtin_amdgcn_tensor_load_to_lds(g0, g1, z4, z4, z8, 0);
}
#endif

// ---------------------------------------------------------------------------
// Elementwise f32 -> bf16 convert
// ---------------------------------------------------------------------------
__global__ __launch_bounds__(256) void cvt_f32_bf16(const float* __restrict__ in,
                                                    unsigned short* __restrict__ out,
                                                    size_t n) {
    size_t i = (size_t)blockIdx.x * 256 + threadIdx.x;
    if (i < n) out[i] = f32_to_bf16(in[i]);
}

// ---------------------------------------------------------------------------
// RMSNorm: one block per row, f32 in (row stride stride_in, first `len` cols),
// bf16 out (dense rows of length len)
// ---------------------------------------------------------------------------
__global__ __launch_bounds__(256) void rmsnorm_bf16(const float* __restrict__ x,
                                                    const float* __restrict__ w,
                                                    unsigned short* __restrict__ out,
                                                    int len, int stride_in) {
    __shared__ float red[256];
    const int row = blockIdx.x;
    const float* xr = x + (size_t)row * stride_in;
    float ss = 0.0f;
    for (int i = threadIdx.x; i < len; i += 256) { float v = xr[i]; ss += v * v; }
    red[threadIdx.x] = ss;
    __syncthreads();
    for (int s = 128; s > 0; s >>= 1) {
        if ((int)threadIdx.x < s) red[threadIdx.x] += red[threadIdx.x + s];
        __syncthreads();
    }
    const float inv = rsqrtf(red[0] / (float)len + 1e-6f);
    for (int i = threadIdx.x; i < len; i += 256)
        out[(size_t)row * len + i] = f32_to_bf16(xr[i] * inv * w[i]);
}

// ---------------------------------------------------------------------------
// Generic bf16 x bf16 -> f32 GEMM. C[M,N] = A[M,K] @ B[K,N].
// Block tile 64x64, K-step 32, 128 threads = 4 waves.
// A tile streamed by the Tensor Data Mover into double-buffered LDS while
// WMMAs run on the other buffer; B tile staged transposed (BsT[n][k]) so
// B-fragment K-pairs are contiguous dword loads. One barrier per k-step.
// ---------------------------------------------------------------------------
#define GBM 64
#define GBN 64
#define GBK 32

__global__ __launch_bounds__(128) void gemm_bf16_f32(const unsigned short* __restrict__ A,
                                                     const unsigned short* __restrict__ B,
                                                     float* __restrict__ C,
                                                     int M, int N, int K) {
    __shared__ unsigned short As[2][GBM][GBK];   // [m][k]
    __shared__ unsigned short BsT[2][GBN][GBK];  // [n][k]

    const int tid  = threadIdx.x;
    const int lane = tid & 31;
    const int wave = tid >> 5;      // 0..3
    const int g    = lane >> 4;     // half-wave select
    const int l16  = lane & 15;

    const int m0 = blockIdx.y * GBM;
    const int n0 = blockIdx.x * GBN;

    f32x8 acc[4] = {};

    auto loadA = [&](int buf, int k0) {
#if HAVE_TDM
        if (tid < 32)
            tdm_load_2d((unsigned)(unsigned long long)&As[buf][0][0],
                        A + (size_t)m0 * K + k0, GBK, GBM,
                        (unsigned)(K - k0), (unsigned)(M - m0),
                        (unsigned long long)K);
#else
        const int r = tid >> 1;
        const int c = (tid & 1) * 16;
        const unsigned short* src = A + (size_t)(m0 + r) * K + k0 + c;
#pragma unroll
        for (int i = 0; i < 8; ++i)
            *(unsigned int*)&As[buf][r][c + 2 * i] = *(const unsigned int*)(src + 2 * i);
#endif
    };
    auto stageB = [&](int buf, int k0) {
        const int n  = tid & 63;
        const int kk = (tid >> 6) * 16;
        const unsigned short* src = B + (size_t)(k0 + kk) * N + n0 + n;
        __builtin_prefetch(src, 0, 0);
#pragma unroll
        for (int i = 0; i < 16; ++i)
            BsT[buf][n][kk + i] = src[(size_t)i * N];
    };

    // prologue: tile 0 into buffer 0
    loadA(0, 0);
    stageB(0, 0);
#if HAVE_TDM
    if (tid < 32) __builtin_amdgcn_s_wait_tensorcnt(0);
#endif
    __syncthreads();

    const int nk = K / GBK;
    for (int ki = 0; ki < nk; ++ki) {
        const int buf = ki & 1;
        if (ki + 1 < nk) {           // async-prefetch next tile into other buffer
            loadA(buf ^ 1, (ki + 1) * GBK);
            stageB(buf ^ 1, (ki + 1) * GBK);
        }

        // A fragment: 16x32 bf16, ISA layout
        Frag a;
#pragma unroll
        for (int v = 0; v < 8; ++v) {
            const int kk = (v & 3) * 2 + (v >> 2) * 16 + g * 8;
            a.u[v] = *(const unsigned int*)&As[buf][wave * 16 + l16][kk];
        }
#pragma unroll
        for (int t = 0; t < 4; ++t) {
            Frag b;
#pragma unroll
            for (int v = 0; v < 8; ++v)
                b.u[v] = *(const unsigned int*)&BsT[buf][t * 16 + l16][(v & 3) * 2 + (v >> 2) * 16 + g * 8];
            acc[t] = __builtin_amdgcn_wmma_f32_16x16x32_bf16(
                false, a.v, false, b.v, (short)0, acc[t], false, false);
        }

#if HAVE_TDM
        if (tid < 32) __builtin_amdgcn_s_wait_tensorcnt(0);
#endif
        __syncthreads();
    }

    // C/D layout: VGPR r -> M = r + 8*g, N = l16 (+16*t)
#pragma unroll
    for (int t = 0; t < 4; ++t) {
        const int n = n0 + t * 16 + l16;
#pragma unroll
        for (int r = 0; r < 8; ++r) {
            const int m = m0 + wave * 16 + 8 * g + r;
            C[(size_t)m * N + n] = acc[t][r];
        }
    }
}

// ---------------------------------------------------------------------------
// RoPE + pack Q: q_f32[T, H*192] -> q_bf[T,H,192], scaled by llama_4_scaling
// grid (T, H), 192 threads
// ---------------------------------------------------------------------------
__global__ __launch_bounds__(192) void rope_build_q(const float* __restrict__ q,
                                                    const int* __restrict__ pos,
                                                    const float* __restrict__ cs,
                                                    const float* __restrict__ lscale,
                                                    unsigned short* __restrict__ qbf) {
    const int t = blockIdx.x, h = blockIdx.y, d = threadIdx.x;
    const float* qr = q + ((size_t)t * 16 + h) * 192;
    const float sc = lscale[t];
    float val;
    if (d < 128) {
        val = qr[d];
    } else {
        const float* c = cs + (size_t)pos[t] * 64;
        const int i = d - 128;
        if (i < 32) val = qr[128 + i] * c[i] - qr[160 + i] * c[32 + i];
        else { const int j = i - 32; val = qr[160 + j] * c[j] + qr[128 + j] * c[32 + j]; }
    }
    qbf[((size_t)t * 16 + h) * 192 + d] = f32_to_bf16(val * sc);
}

// ---------------------------------------------------------------------------
// Pack K (nope + broadcast roped k_pe) and V: grid (T, H), 192 threads
// ---------------------------------------------------------------------------
__global__ __launch_bounds__(192) void rope_build_kv(const float* __restrict__ kv,
                                                     const float* __restrict__ latent,
                                                     const int* __restrict__ pos,
                                                     const float* __restrict__ cs,
                                                     unsigned short* __restrict__ kbf,
                                                     unsigned short* __restrict__ vbf) {
    const int t = blockIdx.x, h = blockIdx.y, d = threadIdx.x;
    const float* kvr = kv + ((size_t)t * 16 + h) * 256;
    float val;
    if (d < 128) {
        val = kvr[d];
        vbf[((size_t)t * 16 + h) * 128 + d] = f32_to_bf16(kvr[128 + d]);
    } else {
        const float* pe = latent + (size_t)t * 576 + 512;
        const float* c  = cs + (size_t)pos[t] * 64;
        const int i = d - 128;
        if (i < 32) val = pe[i] * c[i] - pe[32 + i] * c[32 + i];
        else { const int j = i - 32; val = pe[32 + j] * c[j] + pe[j] * c[32 + j]; }
    }
    kbf[((size_t)t * 16 + h) * 192 + d] = f32_to_bf16(val);
}

// ---------------------------------------------------------------------------
// Flash attention (causal), bf16 QKV, f32 accumulation, bf16 attn output.
// grid (T/64, H), 128 threads = 4 waves; wave owns 16 query rows.
// K tile (64x192) streamed by TDM; V tile staged transposed manually.
// ---------------------------------------------------------------------------
#define AQ 64
#define AK 64

__global__ __launch_bounds__(128) void mla_flash_attn(const unsigned short* __restrict__ Qb,
                                                      const unsigned short* __restrict__ Kb,
                                                      const unsigned short* __restrict__ Vb,
                                                      unsigned short* __restrict__ Ob) {
    __shared__ unsigned short Ks[AK][192];    // [key][d]   24 KB
    __shared__ unsigned short VsT[128][AK];   // [dv][key]  16 KB
    __shared__ unsigned short Ps[4][16][AK];  // per-wave P  8 KB

    const int tid  = threadIdx.x;
    const int lane = tid & 31;
    const int wave = tid >> 5;
    const int g    = lane >> 4;
    const int l16  = lane & 15;
    const int H    = 16;

    const int h  = blockIdx.y;
    const int q0 = blockIdx.x * AQ;

    // Q fragments: 16 rows x 192 = 6 chunks of 16x32
    Frag qf[6];
    {
        const int m = q0 + wave * 16 + l16;
        const unsigned short* qrow = Qb + ((size_t)m * H + h) * 192;
#pragma unroll
        for (int c = 0; c < 6; ++c)
#pragma unroll
            for (int v = 0; v < 8; ++v) {
                const int kk = c * 32 + (v & 3) * 2 + (v >> 2) * 16 + g * 8;
                qf[c].u[v] = *(const unsigned int*)(qrow + kk);
            }
    }

    f32x8 o[8] = {};
    float rowM[8], rowL[8];
#pragma unroll
    for (int r = 0; r < 8; ++r) { rowM[r] = -3.0e38f; rowL[r] = 0.0f; }

    const float scale = 0.07216878364870323f;  // 1/sqrt(192)
    const int qbase = q0 + wave * 16 + 8 * g;  // +r = this slot's global q row

    const int jmax = blockIdx.x;
    for (int j = 0; j <= jmax; ++j) {
        const int k0 = j * AK;
        // stage K tile (64x192): TDM DMA if available, else manual
#if HAVE_TDM
        if (tid < 32)
            tdm_load_2d((unsigned)(unsigned long long)&Ks[0][0],
                        Kb + ((size_t)k0 * H + h) * 192, 192, AK, 192, AK,
                        (unsigned long long)H * 192);
#else
        {
            const int key = tid >> 1;
            const int c0  = (tid & 1) * 96;
            const unsigned int* src =
                (const unsigned int*)(Kb + ((size_t)(k0 + key) * H + h) * 192 + c0);
            unsigned int* dst = (unsigned int*)&Ks[key][c0];
#pragma unroll
            for (int i = 0; i < 48; ++i) dst[i] = src[i];
        }
#endif
        // stage V tile transposed: VsT[d][key]
        {
            const int key = tid >> 1;
            const int d0  = (tid & 1) * 64;
            const unsigned short* src = Vb + ((size_t)(k0 + key) * H + h) * 128 + d0;
#pragma unroll
            for (int i = 0; i < 64; ++i) VsT[d0 + i][key] = src[i];
        }
#if HAVE_TDM
        if (tid < 32) __builtin_amdgcn_s_wait_tensorcnt(0);
#endif
        __syncthreads();

        // S = Q @ K^T : 4 key tiles x 6 K-chunks of WMMA
        f32x8 s[4] = {};
#pragma unroll
        for (int t = 0; t < 4; ++t)
#pragma unroll
            for (int c = 0; c < 6; ++c) {
                Frag b;
#pragma unroll
                for (int v = 0; v < 8; ++v)
                    b.u[v] = *(const unsigned int*)&Ks[t * 16 + l16][c * 32 + (v & 3) * 2 + (v >> 2) * 16 + g * 8];
                s[t] = __builtin_amdgcn_wmma_f32_16x16x32_bf16(
                    false, qf[c].v, false, b.v, (short)0, s[t], false, false);
            }

        // scale + causal mask + running row max
        float newM[8];
#pragma unroll
        for (int r = 0; r < 8; ++r) newM[r] = rowM[r];
#pragma unroll
        for (int t = 0; t < 4; ++t) {
            const int kcol = k0 + t * 16 + l16;
#pragma unroll
            for (int r = 0; r < 8; ++r) {
                const float val = (kcol <= qbase + r) ? s[t][r] * scale : -3.0e38f;
                s[t][r] = val;
                newM[r] = fmaxf(newM[r], val);
            }
        }
#pragma unroll
        for (int r = 0; r < 8; ++r) {  // reduce over the 16 column lanes
            float v = newM[r];
            v = fmaxf(v, __shfl_xor(v, 1, 32));
            v = fmaxf(v, __shfl_xor(v, 2, 32));
            v = fmaxf(v, __shfl_xor(v, 4, 32));
            v = fmaxf(v, __shfl_xor(v, 8, 32));
            newM[r] = v;
        }

        float corr[8], rsum[8];
#pragma unroll
        for (int r = 0; r < 8; ++r) { corr[r] = __expf(rowM[r] - newM[r]); rsum[r] = 0.0f; }

        // P = exp(S - M); spill to per-wave LDS for C-layout -> A-layout
#pragma unroll
        for (int t = 0; t < 4; ++t)
#pragma unroll
            for (int r = 0; r < 8; ++r) {
                const float p = __expf(s[t][r] - newM[r]);
                rsum[r] += p;
                Ps[wave][r + 8 * g][t * 16 + l16] = f32_to_bf16(p);
            }
#pragma unroll
        for (int r = 0; r < 8; ++r) {
            float v = rsum[r];
            v += __shfl_xor(v, 1, 32);
            v += __shfl_xor(v, 2, 32);
            v += __shfl_xor(v, 4, 32);
            v += __shfl_xor(v, 8, 32);
            rowL[r] = rowL[r] * corr[r] + v;
            rowM[r] = newM[r];
        }
#pragma unroll
        for (int t = 0; t < 8; ++t)
#pragma unroll
            for (int r = 0; r < 8; ++r) o[t][r] *= corr[r];

        // O += P @ V : 2 K-chunks x 8 dv tiles
        Frag pf[2];
#pragma unroll
        for (int c = 0; c < 2; ++c)
#pragma unroll
            for (int v = 0; v < 8; ++v) {
                const int kk = c * 32 + (v & 3) * 2 + (v >> 2) * 16 + g * 8;
                pf[c].u[v] = *(const unsigned int*)&Ps[wave][l16][kk];
            }
#pragma unroll
        for (int t = 0; t < 8; ++t)
#pragma unroll
            for (int c = 0; c < 2; ++c) {
                Frag b;
#pragma unroll
                for (int v = 0; v < 8; ++v)
                    b.u[v] = *(const unsigned int*)&VsT[t * 16 + l16][c * 32 + (v & 3) * 2 + (v >> 2) * 16 + g * 8];
                o[t] = __builtin_amdgcn_wmma_f32_16x16x32_bf16(
                    false, pf[c].v, false, b.v, (short)0, o[t], false, false);
            }
        __syncthreads();
    }

    // normalize + store attn as bf16 [T, H*128]
#pragma unroll
    for (int t = 0; t < 8; ++t) {
        const int dv = t * 16 + l16;
#pragma unroll
        for (int r = 0; r < 8; ++r) {
            const int m = q0 + wave * 16 + 8 * g + r;
            Ob[(size_t)m * (16 * 128) + h * 128 + dv] = f32_to_bf16(o[t][r] / rowL[r]);
        }
    }
}

// ---------------------------------------------------------------------------
// Host orchestration
// ---------------------------------------------------------------------------
extern "C" void kernel_launch(void* const* d_in, const int* in_sizes, int n_in,
                              void* d_out, int out_size, void* d_ws, size_t ws_size,
                              hipStream_t stream) {
    (void)in_sizes; (void)n_in; (void)out_size; (void)ws_size;
    const int   T   = 3072, HID = 5120, H = 16;
    const int   QR  = 1536, KVR = 512;
    const int   DQK = 192,  DNV = 256, DV = 128;

    const int*   positions = (const int*)  d_in[0];
    const float* hidden    = (const float*)d_in[1];
    const float* lscale    = (const float*)d_in[2];
    const float* w_q_a     = (const float*)d_in[3];
    const float* q_a_ln_w  = (const float*)d_in[4];
    const float* w_q_b     = (const float*)d_in[5];
    const float* w_kv_a    = (const float*)d_in[6];
    const float* kv_a_ln_w = (const float*)d_in[7];
    const float* w_kv_b    = (const float*)d_in[8];
    const float* w_o       = (const float*)d_in[9];
    const float* cs_cache  = (const float*)d_in[10];
    float*       out       = (float*)d_out;

    char* ws = (char*)d_ws;
    size_t off = 0;
    auto alloc = [&](size_t bytes) -> char* {
        char* p = ws + off;
        off += bytes;
        off = (off + 255) & ~(size_t)255;
        return p;
    };

    unsigned short* hid_bf  = (unsigned short*)alloc((size_t)T * HID * 2);
    unsigned short* wqa_bf  = (unsigned short*)alloc((size_t)HID * QR * 2);
    unsigned short* wqb_bf  = (unsigned short*)alloc((size_t)QR * H * DQK * 2);
    unsigned short* wkva_bf = (unsigned short*)alloc((size_t)HID * (KVR + 64) * 2);
    unsigned short* wkvb_bf = (unsigned short*)alloc((size_t)KVR * H * DNV * 2);
    unsigned short* wo_bf   = (unsigned short*)alloc((size_t)H * DV * HID * 2);
    float*          qa_f    = (float*)alloc((size_t)T * QR * 4);
    unsigned short* qan_bf  = (unsigned short*)alloc((size_t)T * QR * 2);
    float*          q_f     = (float*)alloc((size_t)T * H * DQK * 4);
    float*          lat_f   = (float*)alloc((size_t)T * (KVR + 64) * 4);
    unsigned short* kvan_bf = (unsigned short*)alloc((size_t)T * KVR * 2);
    float*          kv_f    = (float*)alloc((size_t)T * H * DNV * 4);
    unsigned short* q_bf    = (unsigned short*)alloc((size_t)T * H * DQK * 2);
    unsigned short* k_bf    = (unsigned short*)alloc((size_t)T * H * DQK * 2);
    unsigned short* v_bf    = (unsigned short*)alloc((size_t)T * H * DV * 2);
    unsigned short* attn_bf = (unsigned short*)alloc((size_t)T * H * DV * 2);

    auto cvt = [&](const float* src, unsigned short* dst, size_t n) {
        cvt_f32_bf16<<<dim3((unsigned)((n + 255) / 256)), 256, 0, stream>>>(src, dst, n);
    };
    cvt(hidden, hid_bf,  (size_t)T * HID);
    cvt(w_q_a,  wqa_bf,  (size_t)HID * QR);
    cvt(w_q_b,  wqb_bf,  (size_t)QR * H * DQK);
    cvt(w_kv_a, wkva_bf, (size_t)HID * (KVR + 64));
    cvt(w_kv_b, wkvb_bf, (size_t)KVR * H * DNV);
    cvt(w_o,    wo_bf,   (size_t)H * DV * HID);

    auto gemm = [&](const unsigned short* A, const unsigned short* B, float* C,
                    int M, int N, int K) {
        gemm_bf16_f32<<<dim3(N / GBN, M / GBM), 128, 0, stream>>>(A, B, C, M, N, K);
    };

    // q_a = hidden @ w_q_a ; rmsnorm -> bf16
    gemm(hid_bf, wqa_bf, qa_f, T, QR, HID);
    rmsnorm_bf16<<<T, 256, 0, stream>>>(qa_f, q_a_ln_w, qan_bf, QR, QR);

    // q = qa_n @ w_q_b   (T x 3072)
    gemm(qan_bf, wqb_bf, q_f, T, H * DQK, QR);

    // latent = hidden @ w_kv_a  (T x 576) ; rmsnorm first 512 cols
    gemm(hid_bf, wkva_bf, lat_f, T, KVR + 64, HID);
    rmsnorm_bf16<<<T, 256, 0, stream>>>(lat_f, kv_a_ln_w, kvan_bf, KVR, KVR + 64);

    // kv = kv_a_n @ w_kv_b  (T x 4096)
    gemm(kvan_bf, wkvb_bf, kv_f, T, H * DNV, KVR);

    // RoPE + pack to bf16 Q/K/V
    rope_build_q <<<dim3(T, H), 192, 0, stream>>>(q_f, positions, cs_cache, lscale, q_bf);
    rope_build_kv<<<dim3(T, H), 192, 0, stream>>>(kv_f, lat_f, positions, cs_cache, k_bf, v_bf);

    // causal flash attention -> attn_bf [T, H*128]
    mla_flash_attn<<<dim3(T / AQ, H), 128, 0, stream>>>(q_bf, k_bf, v_bf, attn_bf);

    // out = attn @ w_o  (f32 out)
    gemm(attn_bf, wo_bf, out, T, HID, H * DV);
}
